// cylinder_fea_56160992362888
// MI455X (gfx1250) — matrix operand order repaired and checked by reading, not compiled, not run on hardware
//
#include <hip/hip_runtime.h>
#include <hip/hip_bf16.h>

// ---------------------------------------------------------------------------
// Cylinder3D voxel-feature pipeline for MI455X (gfx1250, wave32, WMMA).
//
//  K1: zero-init workspace (encoded-max buffer + occupancy flags)
//  K2: per-point MLP (9->64->128->256->64, BN folded, f16 WMMA, f32 accum)
//      + scatter-max via monotonic float->u32 encoding + global_atomic_max_u32
//  K3: vox[NS,64] @ w128[64,128] + b128  (f16 WMMA, f32 accum, f32 stores)
//
// Workspace: NS*64 u32 encoded-max (176.9 MB) + NS u32 flags (2.8 MB).
// ---------------------------------------------------------------------------

typedef _Float16 v16h __attribute__((ext_vector_type(16)));
typedef _Float16 h8   __attribute__((ext_vector_type(8)));
typedef float    v8f  __attribute__((ext_vector_type(8)));

#define GXD 240
#define GYD 180
#define GZD 16
#define NSVOX (GXD * GYD * GZD)   // 691200

// ---- monotonic float <-> u32 encoding for atomic max over signed floats ----
__device__ __forceinline__ unsigned fenc(float x) {
  unsigned u = __float_as_uint(x);
  return (u & 0x80000000u) ? ~u : (u | 0x80000000u);
}
__device__ __forceinline__ float fdec(unsigned u) {
  return (u & 0x80000000u) ? __uint_as_float(u ^ 0x80000000u)
                           : __uint_as_float(~u);
}

// ---- WMMA fragment loaders (wave32 layouts per CDNA5 ISA 7.12.2) ----------
// A (16x32 f16): lane l holds row m=l&15; K-half kh=l>>4 selects
//   halves {kh*8..kh*8+7} and {16+kh*8..16+kh*8+7}  -> two aligned 16B loads.
__device__ __forceinline__ v16h load_a_frag(const _Float16* act, int m, int kh,
                                            int kbase, int sp) {
  const _Float16* p = act + m * sp + kbase + kh * 8;
  h8 lo = *(const h8*)(p);
  h8 hi = *(const h8*)(p + 16);
  return __builtin_shufflevector(lo, hi, 0, 1, 2, 3, 4, 5, 6, 7,
                                 8, 9, 10, 11, 12, 13, 14, 15);
}
// B (32x16 f16): pre-staged in fragment-major order -> 16 contiguous halves
// per lane: wf[((kc*nT + nt)*32 + lane)*16 + j]  (lane l: n=l&15, K-half l>>4)
__device__ __forceinline__ v16h load_b_frag(const _Float16* wf, int kc, int nt,
                                            int nT, int lane) {
  const _Float16* p = wf + (((kc * nT + nt) * 32 + lane) << 4);
  h8 lo = *(const h8*)(p);
  h8 hi = *(const h8*)(p + 8);
  return __builtin_shufflevector(lo, hi, 0, 1, 2, 3, 4, 5, 6, 7,
                                 8, 9, 10, 11, 12, 13, 14, 15);
}

// Stage a row-major [K][N] f32 weight matrix into f16 B-fragment order in LDS.
__device__ void stage_wfrag(const float* __restrict__ W, _Float16* dst,
                            int realK, int Kpad, int Nout, int tid, int nth) {
  const int nT = Nout >> 4;
  const int total = Kpad * Nout;
  for (int idx = tid; idx < total; idx += nth) {
    int j    = idx & 15;
    int lane = (idx >> 4) & 31;
    int t    = idx >> 9;              // kc*nT + nt
    int nt   = t % nT;
    int kc   = t / nT;
    int k = kc * 32 + (lane >> 4) * 8 + (j & 7) + (j >> 3) * 16;
    int n = nt * 16 + (lane & 15);
    float v = (k < realK) ? W[k * Nout + n] : 0.0f;
    dst[idx] = (_Float16)v;
  }
}

// One folded MLP layer for a 16-row tile:
//   out = relu((A_tile @ W) * Ac[n] + Bc[n]),   K = kC*32, N = nT*16
__device__ __forceinline__ void do_layer(const _Float16* actIn,
                                         const _Float16* wf, const float* Ac,
                                         const float* Bc, _Float16* actOut,
                                         int kC, int nT, int lane, int sp,
                                         bool relu) {
  const int kh = lane >> 4, nc = lane & 15;
  for (int nt = 0; nt < nT; ++nt) {
    v8f c = {};
    for (int kc = 0; kc < kC; ++kc) {
      v16h a = load_a_frag(actIn, nc, kh, kc << 5, sp);
      v16h b = load_b_frag(wf, kc, nt, nT, lane);
      c = __builtin_amdgcn_wmma_f32_16x16x32_f16(false, a, false, b, (short)0,
                                                 c, false, false);
    }
    const int n = (nt << 4) + nc;
    const float sA = Ac[n], sB = Bc[n];
#pragma unroll
    for (int r = 0; r < 8; ++r) {
      float y = c[r] * sA + sB;          // D layout: m = r + 8*kh, n = lane&15
      if (relu) y = fmaxf(y, 0.0f);
      actOut[(r + (kh << 3)) * sp + n] = (_Float16)y;
    }
  }
}

// ---- LDS budget constants (kernel 2, dynamic LDS; one WG per WGP) ---------
constexpr int WF1_H = 32 * 64;      // 2048 halves
constexpr int WF2_H = 64 * 128;     // 8192
constexpr int WF3_H = 128 * 256;    // 32768
constexpr int WF4_H = 256 * 64;     // 16384
constexpr int WFTOT_H = WF1_H + WF2_H + WF3_H + WF4_H;   // 59392 halves
constexpr int PRM_F = 992;          // folded BN params (floats)
constexpr int SP1 = 264;            // act row stride in halves (256 + 8 pad)
constexpr int ACT_H = 16 * SP1;     // one 16-row activation tile
constexpr int WAVES2 = 8;           // 256 threads
constexpr size_t LDS2_BYTES = (size_t)WFTOT_H * 2 + (size_t)PRM_F * 4 +
                              (size_t)WAVES2 * 2 * ACT_H * 2 +
                              (size_t)WAVES2 * 16 * 4;   // ~258 KB < 320 KB

// ---------------------------------------------------------------------------
__global__ void k_init(unsigned* __restrict__ p, int nWords) {
  int i = blockIdx.x * blockDim.x + threadIdx.x;
  int stride = gridDim.x * blockDim.x;
  for (; i < nWords; i += stride) p[i] = 0u;
}

// ---------------------------------------------------------------------------
__global__ __launch_bounds__(256) void k_mlp_scatter(
    const float* __restrict__ pt_fea, const int* __restrict__ xy_ind,
    const float* __restrict__ bn0_g, const float* __restrict__ bn0_b,
    const float* __restrict__ w1, const float* __restrict__ b1,
    const float* __restrict__ bn1_g, const float* __restrict__ bn1_b,
    const float* __restrict__ w2, const float* __restrict__ b2,
    const float* __restrict__ bn2_g, const float* __restrict__ bn2_b,
    const float* __restrict__ w3, const float* __restrict__ b3,
    const float* __restrict__ bn3_g, const float* __restrict__ bn3_b,
    const float* __restrict__ w4, const float* __restrict__ b4,
    unsigned* __restrict__ voxEnc, unsigned* __restrict__ voxFlag, int Npts,
    int nTiles) {
  extern __shared__ unsigned char smem[];
  _Float16* wf1 = (_Float16*)smem;
  _Float16* wf2 = wf1 + WF1_H;
  _Float16* wf3 = wf2 + WF2_H;
  _Float16* wf4 = wf3 + WF3_H;
  float* prm = (float*)(wf4 + WF4_H);
  _Float16* actBase = (_Float16*)(prm + PRM_F);
  int* idBase = (int*)(actBase + WAVES2 * 2 * ACT_H);

  const int tid = threadIdx.x, nth = blockDim.x;
  const float RS = rsqrtf(1.0f + 1e-5f);   // BN eval: g / sqrt(var + eps)

  stage_wfrag(w1, wf1, 9, 32, 64, tid, nth);
  stage_wfrag(w2, wf2, 64, 64, 128, tid, nth);
  stage_wfrag(w3, wf3, 128, 128, 256, tid, nth);
  stage_wfrag(w4, wf4, 256, 256, 64, tid, nth);
  // Folded per-column params: y = acc * A[n] + B[n],  B = b_lin*A + bn_b
  for (int i = tid; i < 16; i += nth) {
    prm[i]      = (i < 9) ? bn0_g[i] * RS : 0.0f;
    prm[16 + i] = (i < 9) ? bn0_b[i] : 0.0f;
  }
  float* A1 = prm + 32;  float* B1 = prm + 96;
  float* A2 = prm + 160; float* B2 = prm + 288;
  float* A3 = prm + 416; float* B3 = prm + 672;
  float* B4 = prm + 928;
  for (int i = tid; i < 64; i += nth) {
    float s = bn1_g[i] * RS; A1[i] = s; B1[i] = b1[i] * s + bn1_b[i];
  }
  for (int i = tid; i < 128; i += nth) {
    float s = bn2_g[i] * RS; A2[i] = s; B2[i] = b2[i] * s + bn2_b[i];
  }
  for (int i = tid; i < 256; i += nth) {
    float s = bn3_g[i] * RS; A3[i] = s; B3[i] = b3[i] * s + bn3_b[i];
  }
  for (int i = tid; i < 64; i += nth) B4[i] = b4[i];
  __syncthreads();

  const int wave = tid >> 5, lane = tid & 31;
  const int kh = lane >> 4, nc = lane & 15;
  _Float16* actA = actBase + wave * (2 * ACT_H);
  _Float16* actB = actA + ACT_H;
  int* ids = idBase + wave * 16;
  const int wavesPerBlk = nth >> 5;
  const int gstride = gridDim.x * wavesPerBlk;

  for (int tile = blockIdx.x * wavesPerBlk + wave; tile < nTiles;
       tile += gstride) {
    const int base = tile << 4;
    // Stage 16x9 input (BN0 applied), zero-padded to K=32, as f16.
    for (int idx = lane; idx < 16 * 32; idx += 32) {
      int m = idx >> 5, f = idx & 31;
      int p = base + m;
      float v = 0.0f;
      if (p < Npts && f < 9) v = pt_fea[p * 9 + f] * prm[f] + prm[16 + f];
      actA[m * SP1 + f] = (_Float16)v;
    }
    if (lane < 16) {
      int p = base + lane, id = -1;
      if (p < Npts) {
        id = (xy_ind[p * 3 + 0] * GYD + xy_ind[p * 3 + 1]) * GZD +
             xy_ind[p * 3 + 2];
        voxFlag[id] = 1u;      // racy-but-idempotent occupancy flag
      }
      ids[lane] = id;
    }
    // same-wave LDS ops are in-order -> no barrier needed between layers
    do_layer(actA, wf1, A1, B1, actB, 1, 4, lane, SP1, true);    //   9-> 64
    do_layer(actB, wf2, A2, B2, actA, 2, 8, lane, SP1, true);    //  64->128
    do_layer(actA, wf3, A3, B3, actB, 4, 16, lane, SP1, true);   // 128->256
    // Layer 4 (256->64, no BN/ReLU) fused with scatter-max.
    for (int nt = 0; nt < 4; ++nt) {
      v8f c = {};
      for (int kc = 0; kc < 8; ++kc) {
        v16h a = load_a_frag(actB, nc, kh, kc << 5, SP1);
        v16h b = load_b_frag(wf4, kc, nt, 4, lane);
        c = __builtin_amdgcn_wmma_f32_16x16x32_f16(false, a, false, b,
                                                   (short)0, c, false, false);
      }
      const int n = (nt << 4) + nc;
      const float bb = B4[n];
#pragma unroll
      for (int r = 0; r < 8; ++r) {
        int id = ids[r + (kh << 3)];
        if (id >= 0)
          atomicMax(voxEnc + (size_t)id * 64 + n, fenc(c[r] + bb));
      }
    }
  }
}

// ---------------------------------------------------------------------------
constexpr int SP3 = 72;  // 64 + 8 halves padding

__global__ __launch_bounds__(256) void k_voxgemm(
    const unsigned* __restrict__ voxEnc, const unsigned* __restrict__ voxFlag,
    const float* __restrict__ w128, const float* __restrict__ b128,
    float* __restrict__ out, int nTiles) {
  __shared__ __align__(16) _Float16 wf[64 * 128];
  __shared__ float bcol[128];
  __shared__ __align__(16) _Float16 act[8][16 * SP3];
  __shared__ int fbuf[8][16];

  const int tid = threadIdx.x, nth = blockDim.x;
  stage_wfrag(w128, wf, 64, 64, 128, tid, nth);
  for (int i = tid; i < 128; i += nth) bcol[i] = b128[i];
  __syncthreads();

  const int wave = tid >> 5, lane = tid & 31;
  const int kh = lane >> 4, nc = lane & 15;
  const int wavesPerBlk = nth >> 5;
  const int gstride = gridDim.x * wavesPerBlk;

  for (int tile = blockIdx.x * wavesPerBlk + wave; tile < nTiles;
       tile += gstride) {
    const int base = tile << 4;
    if (lane < 16) fbuf[wave][lane] = (int)voxFlag[base + lane];
    for (int idx = lane; idx < 16 * 64; idx += 32) {
      int m = idx >> 6, k = idx & 63;
      float v =
          fbuf[wave][m] ? fdec(voxEnc[(size_t)(base + m) * 64 + k]) : 0.0f;
      act[wave][m * SP3 + k] = (_Float16)v;
    }
    for (int nt = 0; nt < 8; ++nt) {
      v8f c = {};
      for (int kc = 0; kc < 2; ++kc) {
        v16h a = load_a_frag(act[wave], nc, kh, kc << 5, SP3);
        v16h b = load_b_frag(wf, kc, nt, 8, lane);
        c = __builtin_amdgcn_wmma_f32_16x16x32_f16(false, a, false, b,
                                                   (short)0, c, false, false);
      }
      const int n = (nt << 4) + nc;
      const float bb = bcol[n];
#pragma unroll
      for (int r = 0; r < 8; ++r) {
        out[(size_t)(base + r + (kh << 3)) * 128 + n] = c[r] + bb;
      }
    }
  }
}

// ---------------------------------------------------------------------------
extern "C" void kernel_launch(void* const* d_in, const int* in_sizes, int n_in,
                              void* d_out, int out_size, void* d_ws,
                              size_t ws_size, hipStream_t stream) {
  const float* pt_fea = (const float*)d_in[0];
  const int* xy_ind   = (const int*)d_in[1];
  const float* bn0_g  = (const float*)d_in[2];
  const float* bn0_b  = (const float*)d_in[3];
  const float* w1 = (const float*)d_in[4];
  const float* b1 = (const float*)d_in[5];
  const float* bn1_g = (const float*)d_in[6];
  const float* bn1_b = (const float*)d_in[7];
  const float* w2 = (const float*)d_in[8];
  const float* b2 = (const float*)d_in[9];
  const float* bn2_g = (const float*)d_in[10];
  const float* bn2_b = (const float*)d_in[11];
  const float* w3 = (const float*)d_in[12];
  const float* b3 = (const float*)d_in[13];
  const float* bn3_g = (const float*)d_in[14];
  const float* bn3_b = (const float*)d_in[15];
  const float* w4 = (const float*)d_in[16];
  const float* b4 = (const float*)d_in[17];
  const float* w128 = (const float*)d_in[18];
  const float* b128 = (const float*)d_in[19];

  const int Npts = in_sizes[0] / 9;
  unsigned* voxEnc  = (unsigned*)d_ws;
  unsigned* voxFlag = voxEnc + (size_t)NSVOX * 64;

  // K1: zero encoded-max buffer (enc==0 is below every real encoding) + flags
  k_init<<<2048, 256, 0, stream>>>(voxEnc, NSVOX * 65);

  // K2: MLP + scatter-max. 8 waves/block, ~258 KB dynamic LDS (1 WG/WGP).
  const int nTiles = (Npts + 15) / 16;
  int blocks2 = (nTiles + WAVES2 - 1) / WAVES2;
  if (blocks2 > 1024) blocks2 = 1024;
  k_mlp_scatter<<<blocks2, 256, LDS2_BYTES, stream>>>(
      pt_fea, xy_ind, bn0_g, bn0_b, w1, b1, bn1_g, bn1_b, w2, b2, bn2_g,
      bn2_b, w3, b3, bn3_g, bn3_b, w4, b4, voxEnc, voxFlag, Npts, nTiles);

  // K3: voxel GEMM -> out[NS,128]. Empty voxels produce b128 (0 @ W + b).
  const int nVTiles = NSVOX / 16;                  // 43200
  int blocks3 = (nVTiles + WAVES2 - 1) / WAVES2;   // 5400
  k_voxgemm<<<blocks3, 256, 0, stream>>>(voxEnc, voxFlag, w128, b128,
                                         (float*)d_out, nVTiles);
}
